// Res_up_62947040690580
// MI455X (gfx1250) — matrix operand order repaired and checked by reading, not compiled, not run on hardware
//
#include <hip/hip_runtime.h>
#include <hip/hip_bf16.h>
#include <stdint.h>

// ---------------------------------------------------------------------------
// CDNA5 (gfx1250, wave32) EdgeConv pipeline using v_wmma_f32_16x16x32_bf16.
// 32-edge tiles (2 M-tiles) per block iteration; weights register-resident as
// WMMA B fragments; LayerNorm via cross-lane shuffle reduction from registers;
// scatter (segment_sum) straight from registers with f32 global atomics.
// ---------------------------------------------------------------------------

typedef __attribute__((ext_vector_type(16))) __bf16 v16bf;
typedef __attribute__((ext_vector_type(8)))  float  v8f;

union Frag {
  v16bf v;
  uint4 q[2];
  unsigned short u[16];
};

__device__ __forceinline__ unsigned short f2bf(float f) {
  unsigned int u = __float_as_uint(f);
  u += 0x7FFFu + ((u >> 16) & 1u);          // round-to-nearest-even
  return (unsigned short)(u >> 16);
}

__device__ __forceinline__ unsigned pack2bf(float a, float b) {
  return ((unsigned)f2bf(b) << 16) | (unsigned)f2bf(a);
}

// --------------------------- helper kernels --------------------------------

__global__ void zero_kernel(float4* __restrict__ p, size_t n4) {
  size_t t = (size_t)blockIdx.x * blockDim.x + threadIdx.x;
  if (t < n4) p[t] = make_float4(0.f, 0.f, 0.f, 0.f);
}

__global__ void leaky_kernel(float4* __restrict__ p, size_t n4) {
  size_t t = (size_t)blockIdx.x * blockDim.x + threadIdx.x;
  if (t < n4) {
    float4 v = p[t];
    v.x = v.x > 0.f ? v.x : 0.01f * v.x;
    v.y = v.y > 0.f ? v.y : 0.01f * v.y;
    v.z = v.z > 0.f ? v.z : 0.01f * v.z;
    v.w = v.w > 0.f ? v.w : 0.01f * v.w;
    p[t] = v;
  }
}

template <int C>  // C multiple of 4
__global__ void unpool_kernel(const float4* __restrict__ src,
                              const int* __restrict__ idx,
                              float4* __restrict__ dst,
                              int Bn, int Nc, int Nf) {
  constexpr int C4 = C / 4;
  size_t t = (size_t)blockIdx.x * blockDim.x + threadIdx.x;
  size_t total = (size_t)Bn * Nc * C4;
  if (t >= total) return;
  int    c  = (int)(t % C4);
  size_t rc = t / C4;
  int    i  = (int)(rc % Nc);
  int    b  = (int)(rc / Nc);
  dst[((size_t)b * Nf + idx[i]) * C4 + c] = src[t];
}

// ------------------------------ EdgeConv ------------------------------------

template <int CIN, int COUT>
__global__ __launch_bounds__(COUT * 2)
void edgeconv_kernel(const float* __restrict__ x,
                     const int* __restrict__ ei, int E, int Nnodes, int Bn,
                     const float* __restrict__ W1, const float* __restrict__ b1,
                     const float* __restrict__ W2, const float* __restrict__ b2,
                     const float* __restrict__ g,  const float* __restrict__ bt,
                     float* __restrict__ acc) {
  constexpr int K1    = 2 * CIN;       // layer-1 K extent
  constexpr int K1_4  = K1 / 4;
  constexpr int NW    = COUT / 16;     // waves per block
  constexpr int BLOCK = 32 * NW;
  constexpr int K1P   = K1 + 8;        // bf16 row stride (16B aligned)
  constexpr int CP    = COUT + 8;

  __shared__ __align__(16) unsigned short featA[32][K1P]; // layer-1 A tiles
  __shared__ __align__(16) unsigned short h1[32][CP];     // layer-2 A tiles
  __shared__ int   ssrc[32], sdst[32];
  __shared__ float smu[32], srs[32];
  __shared__ float pSum[NW][32], pSq[NW][32];

  const int tid    = threadIdx.x;
  const int lane   = tid & 31;
  const int wv     = tid >> 5;
  const int m      = lane & 15;
  const int hi     = lane >> 4;       // 0 | 1
  const int koff8  = hi << 3;         // A-fragment K offset
  const int koff16 = hi << 4;         // unused (B fragments use koff8)
  const int colg   = wv * 16 + m;     // this lane's output column

  // ---- per-lane column params ----
  const float b1v = b1[colg];
  const float b2v = b2[colg];
  const float gv  = g[colg];
  const float btv = bt[colg];

  // ---- load weight fragments (WMMA B layout, bf16) into registers ----
  Frag w1f[K1 / 32];
#pragma unroll
  for (int kt = 0; kt < K1 / 32; ++kt) {
    const int kb = kt * 32 + koff8;
#pragma unroll
    for (int i = 0; i < 16; ++i)
      w1f[kt].u[i] = f2bf(W1[(size_t)(kb + i + (i & 8)) * COUT + colg]);
  }
  Frag w2f[COUT / 32];
#pragma unroll
  for (int kt = 0; kt < COUT / 32; ++kt) {
    const int kb = kt * 32 + koff8;
#pragma unroll
    for (int i = 0; i < 16; ++i)
      w2f[kt].u[i] = f2bf(W2[(size_t)(kb + i + (i & 8)) * COUT + colg]);
  }

  const int tilesPerBatch = (E + 31) >> 5;   // 32 edges per tile
  const int totalTiles    = Bn * tilesPerBatch;

  for (int tile = blockIdx.x; tile < totalTiles; tile += gridDim.x) {
    const int bidx = tile / tilesPerBatch;
    const int e0   = (tile - bidx * tilesPerBatch) << 5;

    __syncthreads();   // protect LDS reuse from previous iteration
    if (tid < 32) {
      int e = e0 + tid; if (e >= E) e = E - 1;
      ssrc[tid] = ei[e];
      sdst[tid] = ei[E + e];
    }
    __syncthreads();

    // ---- gather edge-feature tiles (32 x K1) into LDS as bf16 (float4) ----
    const float* xb = x + (size_t)bidx * Nnodes * CIN;
    for (int lin = tid; lin < 32 * K1_4; lin += BLOCK) {
      const int r  = lin / K1_4;
      const int c4 = (lin - r * K1_4) << 2;
      const float4* pd = (const float4*)(xb + (size_t)sdst[r] * CIN);
      float4 v;
      if (c4 < CIN) {
        v = pd[c4 >> 2];
      } else {
        const float4* ps = (const float4*)(xb + (size_t)ssrc[r] * CIN);
        const int cc = (c4 - CIN) >> 2;
        float4 a = ps[cc], d = pd[cc];
        v = make_float4(a.x - d.x, a.y - d.y, a.z - d.z, a.w - d.w);
      }
      uint2 pk;
      pk.x = pack2bf(v.x, v.y);
      pk.y = pack2bf(v.z, v.w);
      *(uint2*)&featA[r][c4] = pk;
    }
    __syncthreads();

    // ---- layer 1: relu(feat @ W1 + b1) -> h1 (bf16 LDS), 2 M-tiles ----
    v8f acc1a = {0.f, 0.f, 0.f, 0.f, 0.f, 0.f, 0.f, 0.f};
    v8f acc1b = {0.f, 0.f, 0.f, 0.f, 0.f, 0.f, 0.f, 0.f};
#pragma unroll
    for (int kt = 0; kt < K1 / 32; ++kt) {
      Frag a0, a1;
      a0.q[0] = *(const uint4*)&featA[m][kt * 32 + koff8];
      a0.q[1] = *(const uint4*)&featA[m][kt * 32 + 16 + koff8];
      a1.q[0] = *(const uint4*)&featA[m + 16][kt * 32 + koff8];
      a1.q[1] = *(const uint4*)&featA[m + 16][kt * 32 + 16 + koff8];
      acc1a = __builtin_amdgcn_wmma_f32_16x16x32_bf16(false, a0.v, false, w1f[kt].v,
                                                      (short)0, acc1a, false, false);
      acc1b = __builtin_amdgcn_wmma_f32_16x16x32_bf16(false, a1.v, false, w1f[kt].v,
                                                      (short)0, acc1b, false, false);
    }
#pragma unroll
    for (int r = 0; r < 8; ++r) {
      const int row = r + (hi << 3);
      float va = acc1a[r] + b1v;
      float vb = acc1b[r] + b1v;
      h1[row][colg]      = f2bf(va > 0.f ? va : 0.f);
      h1[row + 16][colg] = f2bf(vb > 0.f ? vb : 0.f);
    }
    __syncthreads();

    // ---- layer 2: h1 @ W2 + b2 (accumulators stay in registers) ----
    v8f acc2a = {0.f, 0.f, 0.f, 0.f, 0.f, 0.f, 0.f, 0.f};
    v8f acc2b = {0.f, 0.f, 0.f, 0.f, 0.f, 0.f, 0.f, 0.f};
#pragma unroll
    for (int kt = 0; kt < COUT / 32; ++kt) {
      Frag a0, a1;
      a0.q[0] = *(const uint4*)&h1[m][kt * 32 + koff8];
      a0.q[1] = *(const uint4*)&h1[m][kt * 32 + 16 + koff8];
      a1.q[0] = *(const uint4*)&h1[m + 16][kt * 32 + koff8];
      a1.q[1] = *(const uint4*)&h1[m + 16][kt * 32 + 16 + koff8];
      acc2a = __builtin_amdgcn_wmma_f32_16x16x32_bf16(false, a0.v, false, w2f[kt].v,
                                                      (short)0, acc2a, false, false);
      acc2b = __builtin_amdgcn_wmma_f32_16x16x32_bf16(false, a1.v, false, w2f[kt].v,
                                                      (short)0, acc2b, false, false);
    }
    float vala[8], valb[8];
#pragma unroll
    for (int r = 0; r < 8; ++r) {
      vala[r] = acc2a[r] + b2v;
      valb[r] = acc2b[r] + b2v;
    }

    // ---- LayerNorm partial sums: reduce over 16 columns via shuffles ----
    // Lanes 0-15 of the wave hold row (r+8*0); lanes 16-31 hold row (r+8*1);
    // xor masks < 16 keep the two halves independent.
#pragma unroll
    for (int r = 0; r < 8; ++r) {
      float s0 = vala[r], q0 = s0 * s0;
      float s1 = valb[r], q1 = s1 * s1;
#pragma unroll
      for (int d = 1; d < 16; d <<= 1) {
        s0 += __shfl_xor(s0, d, 32);
        q0 += __shfl_xor(q0, d, 32);
        s1 += __shfl_xor(s1, d, 32);
        q1 += __shfl_xor(q1, d, 32);
      }
      if (m == 0) {
        const int row = r + (hi << 3);
        pSum[wv][row]      = s0;
        pSq[wv][row]       = q0;
        pSum[wv][row + 16] = s1;
        pSq[wv][row + 16]  = q1;
      }
    }
    __syncthreads();

    if (tid < 32) {
      float s = 0.f, q = 0.f;
#pragma unroll
      for (int w = 0; w < NW; ++w) { s += pSum[w][tid]; q += pSq[w][tid]; }
      const float mu  = s / COUT;
      const float var = q / COUT - mu * mu;
      smu[tid] = mu;
      srs[tid] = rsqrtf(var + 1e-5f);
    }
    __syncthreads();

    // ---- normalize + affine + segment_sum scatter, straight from regs ----
    float* accb = acc + (size_t)bidx * Nnodes * COUT;
#pragma unroll
    for (int r = 0; r < 8; ++r) {
      const int row0 = r + (hi << 3);
      const int row1 = row0 + 16;
      const int ea = e0 + row0;
      const int eb = e0 + row1;
      if (ea < E) {
        const float v = (vala[r] - smu[row0]) * srs[row0] * gv + btv;
        atomicAdd(&accb[(size_t)sdst[row0] * COUT + colg], v);
      }
      if (eb < E) {
        const float v = (valb[r] - smu[row1]) * srs[row1] * gv + btv;
        atomicAdd(&accb[(size_t)sdst[row1] * COUT + colg], v);
      }
    }
  }
}

// ------------------------------ launcher ------------------------------------

extern "C" void kernel_launch(void* const* d_in, const int* in_sizes, int n_in,
                              void* d_out, int out_size, void* d_ws, size_t ws_size,
                              hipStream_t stream) {
  (void)in_sizes; (void)n_in; (void)out_size; (void)ws_size;
  const int B = 2, N_C = 16384, N_F = 65536, C = 128;
  const int E_C = 65536, E_F = 262144;

  const float* x    = (const float*)d_in[0];
  const int*   idx  = (const int*)d_in[1];
  const int*   ei_c = (const int*)d_in[2];
  const int*   ei_f = (const int*)d_in[3];
  const float* sW1 = (const float*)d_in[4];
  const float* sb1 = (const float*)d_in[5];
  const float* sW2 = (const float*)d_in[6];
  const float* sb2 = (const float*)d_in[7];
  const float* sg  = (const float*)d_in[8];
  const float* sbt = (const float*)d_in[9];
  const float* aW1 = (const float*)d_in[10];
  const float* ab1 = (const float*)d_in[11];
  const float* aW2 = (const float*)d_in[12];
  const float* ab2 = (const float*)d_in[13];
  const float* ag  = (const float*)d_in[14];
  const float* abt = (const float*)d_in[15];
  const float* bW1 = (const float*)d_in[16];
  const float* bb1 = (const float*)d_in[17];
  const float* bW2 = (const float*)d_in[18];
  const float* bb2 = (const float*)d_in[19];
  const float* bg  = (const float*)d_in[20];
  const float* bbt = (const float*)d_in[21];

  float* out = (float*)d_out;

  // workspace: unpooled x (B*N_F*128), coarse h (B*N_C*64), unpooled h (B*N_F*64)
  float* xun = (float*)d_ws;
  float* hc  = xun + (size_t)B * N_F * C;
  float* hun = hc + (size_t)B * N_C * (C / 2);

  const size_t nOut = (size_t)B * N_F * C;
  const size_t nHc  = (size_t)B * N_C * (C / 2);
  const size_t nHun = (size_t)B * N_F * (C / 2);

  auto gz = [](size_t n) { return (unsigned)((n + 255) / 256); };

  zero_kernel<<<gz(nOut / 4), 256, 0, stream>>>((float4*)out, nOut / 4);
  zero_kernel<<<gz(nOut / 4), 256, 0, stream>>>((float4*)xun, nOut / 4);
  zero_kernel<<<gz(nHc / 4),  256, 0, stream>>>((float4*)hc,  nHc / 4);
  zero_kernel<<<gz(nHun / 4), 256, 0, stream>>>((float4*)hun, nHun / 4);

  // skip = edgeconv(unpool(x), ei_f, s*)  -> accumulate into out
  unpool_kernel<128><<<gz((size_t)B * N_C * 32), 256, 0, stream>>>(
      (const float4*)x, idx, (float4*)xun, B, N_C, N_F);
  edgeconv_kernel<128, 128><<<4096, 256, 0, stream>>>(
      xun, ei_f, E_F, N_F, B, sW1, sb1, sW2, sb2, sg, sbt, out);

  // h = edgeconv(x, ei_c, a*) -> hc (COUT=64)
  edgeconv_kernel<128, 64><<<2048, 128, 0, stream>>>(
      x, ei_c, E_C, N_C, B, aW1, ab1, aW2, ab2, ag, abt, hc);

  // h = edgeconv(unpool(h), ei_f, b*) -> accumulate into out
  unpool_kernel<64><<<gz((size_t)B * N_C * 16), 256, 0, stream>>>(
      (const float4*)hc, idx, (float4*)hun, B, N_C, N_F);
  edgeconv_kernel<64, 128><<<4096, 256, 0, stream>>>(
      hun, ei_f, E_F, N_F, B, bW1, bb1, bW2, bb2, bg, bbt, out);

  // out = leaky_relu(h + skip)
  leaky_kernel<<<gz(nOut / 4), 256, 0, stream>>>((float4*)out, nOut / 4);
}
